// GraphStructureLearner_45423574123046
// MI455X (gfx1250) — matrix operand order repaired and checked by reading, not compiled, not run on hardware
//
#include <hip/hip_runtime.h>
#include <hip/hip_bf16.h>

typedef __attribute__((ext_vector_type(16))) _Float16 v16h;
typedef __attribute__((ext_vector_type(8)))  _Float16 v8h;
typedef __attribute__((ext_vector_type(8)))  float    v8f;

#define GB    16
#define GN    2048
#define GIN   12
#define GH    64
#define GK    16
#define NEG_FLT_MAX (-3.4028234663852886e+38f)

// ---------------------------------------------------------------------------
// Kernel 1: NodeEncoder  emb = relu(x@W1+b1)@W2+b2, plus L2-normalized f16
// copy for the WMMA similarity GEMM.  One thread per node; weights in LDS.
// ---------------------------------------------------------------------------
__global__ __launch_bounds__(256) void gsl_encoder(
    const float* __restrict__ x,  const float* __restrict__ W1,
    const float* __restrict__ b1, const float* __restrict__ W2,
    const float* __restrict__ b2, float* __restrict__ embOut,
    _Float16* __restrict__ nrmOut)
{
  __shared__ float sW1[GIN*GH];
  __shared__ float sB1[GH];
  __shared__ float sW2[GH*GH];
  __shared__ float sB2[GH];
  for (int i = threadIdx.x; i < GIN*GH; i += 256) sW1[i] = W1[i];
  for (int i = threadIdx.x; i < GH*GH;  i += 256) sW2[i] = W2[i];
  if (threadIdx.x < GH) { sB1[threadIdx.x] = b1[threadIdx.x];
                          sB2[threadIdx.x] = b2[threadIdx.x]; }
  __syncthreads();

  const int node = blockIdx.x * 256 + threadIdx.x;      // < B*N
  const float* xp = x + (size_t)node * GIN;
  float xv[GIN];
  #pragma unroll
  for (int t = 0; t < GIN; ++t) xv[t] = xp[t];

  float acc[GH];
  #pragma unroll
  for (int j = 0; j < GH; ++j) acc[j] = sB2[j];
  for (int k = 0; k < GH; ++k) {
    float hk = sB1[k];
    #pragma unroll
    for (int t = 0; t < GIN; ++t) hk += xv[t] * sW1[t*GH + k];
    hk = fmaxf(hk, 0.f);
    const float* w2r = sW2 + k*GH;
    #pragma unroll
    for (int j = 0; j < GH; ++j) acc[j] += hk * w2r[j];
  }

  float ss = 0.f;
  #pragma unroll
  for (int j = 0; j < GH; ++j) ss += acc[j]*acc[j];
  const float inv = 1.f / fmaxf(sqrtf(ss), 1e-12f);

  float*    eo = embOut + (size_t)node * GH;
  _Float16* no = nrmOut + (size_t)node * GH;
  #pragma unroll
  for (int j = 0; j < GH; ++j) { eo[j] = acc[j]; no[j] = (_Float16)(acc[j]*inv); }
}

// ---------------------------------------------------------------------------
// Kernel 2: fused  sim = norm @ normT  (f16 WMMA, f32 accum)  +  top-16 mask.
// One workgroup (8 waves) owns a 16-row stripe: the full 16x2048 sim stripe
// lives in LDS (128 KB of the WGP's 320 KB), so unmasked sim never touches
// HBM -- only the masked adj (268 MB) is written once.
// ---------------------------------------------------------------------------
extern __shared__ char dynsmem[];

__global__ __launch_bounds__(256) void gsl_sim_topk(
    const _Float16* __restrict__ nrm, float* __restrict__ adj)
{
  float* srow = (float*)dynsmem;          // [16][2048] sim stripe
  float* tks  = srow + 16*GN;             // [256][16]  per-lane top-16 spill

  const int b    = blockIdx.y;
  const int i0   = blockIdx.x * 16;
  const int lane = threadIdx.x & 31;
  const int wave = threadIdx.x >> 5;
  const int l15  = lane & 15;
  const int hi   = lane >> 4;

  // ---- A fragments: rows i0..i0+15, K = 0..63 (two 16x16x32 fragments) ----
  // A k order chosen here: lane L = row L%16; low half-wave holds
  // K 0..15, high half-wave K 16..31 (contiguous, matches B).
  const _Float16* Ap = nrm + ((size_t)b*GN + i0 + l15) * GH;
  v16h a0, a1;
  {
    v8h lo0 = *(const v8h*)(Ap +      (hi ? 16 :  0));
    v8h hi0 = *(const v8h*)(Ap +      (hi ? 24 :  8));
    v8h lo1 = *(const v8h*)(Ap + 32 + (hi ? 16 :  0));
    v8h hi1 = *(const v8h*)(Ap + 32 + (hi ? 24 :  8));
    #pragma unroll
    for (int e = 0; e < 8; ++e) {
      a0[e] = lo0[e]; a0[e+8] = hi0[e];
      a1[e] = lo1[e]; a1[e+8] = hi1[e];
    }
  }

  // ---- phase 1: each wave computes 16 of the 128 j-tiles, A reused ----
  for (int jt = wave; jt < GN/16; jt += 8) {
    const int j0 = jt * 16;
    // B = normT: column j of B is norm row j. Lane L = column L%16,
    // contiguous 16 K-values starting at K = (half?16:0) (+k0).
    const _Float16* Bp = nrm + ((size_t)b*GN + j0 + l15) * GH + (hi ? 16 : 0);
    v16h bf0 = *(const v16h*)(Bp);        // K 0..31 chunk
    v16h bf1 = *(const v16h*)(Bp + 32);   // K 32..63 chunk
    v8f c = {};
    c = __builtin_amdgcn_wmma_f32_16x16x32_f16(false, a0, false, bf0,
                                               (short)0, c, false, false);
    c = __builtin_amdgcn_wmma_f32_16x16x32_f16(false, a1, false, bf1,
                                               (short)0, c, false, false);
    const int col = j0 + l15;
    #pragma unroll
    for (int r = 0; r < 8; ++r)           // C/D: (M = r + 8*hi, N = lane%16)
      srow[(hi*8 + r)*GN + col] = c[r];
  }
  __syncthreads();

  // ---- phase 2: exact top-16 per row, masked coalesced write ----
  float* tp = tks + threadIdx.x * 16;
  for (int rr = 0; rr < 2; ++rr) {
    const int r = wave + rr*8;            // 16 rows / 8 waves
    const float*  row = srow + r*GN;
    const float4* rv  = (const float4*)(row + lane*64);   // 64 elems / lane

    // local sorted top-16 over this lane's contiguous chunk
    float t[16];
    #pragma unroll
    for (int k = 0; k < 16; ++k) t[k] = NEG_FLT_MAX;
    #pragma unroll
    for (int i4 = 0; i4 < 16; ++i4) {
      float4 v4 = rv[i4];
      float vs[4] = {v4.x, v4.y, v4.z, v4.w};
      #pragma unroll
      for (int q = 0; q < 4; ++q) {
        float v = vs[q];
        if (v > t[15]) {
          t[15] = v;
          #pragma unroll
          for (int k = 15; k > 0; --k) {  // single bubble pass re-sorts
            float mx = fmaxf(t[k-1], t[k]);
            float mn = fminf(t[k-1], t[k]);
            t[k-1] = mx; t[k] = mn;
          }
        }
      }
    }
    #pragma unroll
    for (int k = 0; k < 16; ++k) tp[k] = t[k];

    // 16 selection rounds: wave-max + ballot pop -> 16th-largest threshold
    int p = 0;
    float thr = NEG_FLT_MAX;
    for (int s = 0; s < GK; ++s) {
      float cand = (p < 16) ? tp[p] : NEG_FLT_MAX;
      float m = cand;
      #pragma unroll
      for (int off = 16; off > 0; off >>= 1)
        m = fmaxf(m, __shfl_xor(m, off, 32));
      unsigned msk = __builtin_amdgcn_ballot_w32(cand == m);
      if (lane == (__ffs(msk) - 1)) ++p;
      thr = m;
    }

    // tie handling: keep all v>thr, plus first (16 - nGreater) v==thr by index
    int nG = 0, eqLoc = 0;
    #pragma unroll
    for (int i4 = 0; i4 < 16; ++i4) {
      float4 v4 = rv[i4];
      float vs[4] = {v4.x, v4.y, v4.z, v4.w};
      #pragma unroll
      for (int q = 0; q < 4; ++q) {
        nG    += (vs[q] >  thr);
        eqLoc += (vs[q] == thr);
      }
    }
    #pragma unroll
    for (int off = 16; off > 0; off >>= 1) nG += __shfl_xor(nG, off, 32);
    int scan = eqLoc;
    #pragma unroll
    for (int off = 1; off < 32; off <<= 1) {
      int u = __shfl_up(scan, (unsigned)off, 32);
      if (lane >= off) scan += u;
    }
    int eqRank = scan - eqLoc;            // equals-before-me (index order)
    const int eqAllow = GK - nG;

    float4* out = (float4*)(adj + ((size_t)b*GN + i0 + r)*GN + lane*64);
    #pragma unroll
    for (int i4 = 0; i4 < 16; ++i4) {
      float4 v4 = rv[i4];
      float vs[4] = {v4.x, v4.y, v4.z, v4.w};
      float os[4];
      #pragma unroll
      for (int q = 0; q < 4; ++q) {
        float v = vs[q];
        bool keep;
        if (v > thr)        keep = true;
        else if (v == thr) { keep = (eqRank < eqAllow); ++eqRank; }
        else                keep = false;
        os[q] = keep ? v : 0.f;
      }
      out[i4] = make_float4(os[0], os[1], os[2], os[3]);
    }
  }
}

// ---------------------------------------------------------------------------
extern "C" void kernel_launch(void* const* d_in, const int* in_sizes, int n_in,
                              void* d_out, int out_size, void* d_ws, size_t ws_size,
                              hipStream_t stream) {
  (void)in_sizes; (void)n_in; (void)out_size; (void)ws_size;
  const float* x  = (const float*)d_in[0];
  const float* W1 = (const float*)d_in[1];
  const float* b1 = (const float*)d_in[2];
  const float* W2 = (const float*)d_in[3];
  const float* b2 = (const float*)d_in[4];

  float* adj    = (float*)d_out;                       // (B,N,N)
  float* embOut = adj + (size_t)GB*GN*GN;              // (B,N,H) raw emb
  _Float16* nrm = (_Float16*)d_ws;                     // (B,N,H) f16 normalized

  gsl_encoder<<<(GB*GN)/256, 256, 0, stream>>>(x, W1, b1, W2, b2, embOut, nrm);

  const size_t ldsBytes = (size_t)(16*GN + 256*16) * sizeof(float); // 144 KB
  gsl_sim_topk<<<dim3(GN/16, GB), 256, ldsBytes, stream>>>(nrm, adj);
}
